// MVQVAE_Transformer_11922829213996
// MI455X (gfx1250) — matrix operand order repaired and checked by reading, not compiled, not run on hardware
//
#include <hip/hip_runtime.h>
#include <stdint.h>

// ---------------- model constants ----------------
#define HIDDEN 1152
#define HEADS  16
#define DH     72
#define DHP    96          // Dh padded to multiple of 32 for WMMA K
#define NPTS   282
#define NPAD   288         // tokens padded to multiple of 16/32
#define MLPH   4608
#define BQ     64
#define MTOK   (BQ * NPTS) // 18048, multiple of 32
#define DCB    64
#define KCB    512
#define QKVN   3456

typedef __attribute__((ext_vector_type(16))) __bf16 bf16x16;
typedef __attribute__((ext_vector_type(8)))  float  floatx8;

union FragB { bf16x16 v; uint4 u[2]; unsigned short h[16]; };

__device__ __forceinline__ unsigned short f2bf(float f) {
  unsigned int u = __float_as_uint(f);
  u += 0x7FFFu + ((u >> 16) & 1u);          // round-to-nearest-even
  return (unsigned short)(u >> 16);
}
__device__ __forceinline__ float bf2f(unsigned short h) {
  return __uint_as_float(((unsigned int)h) << 16);
}

// ---------------- generic bf16 WMMA GEMM ----------------
// A: bf16 [M,K] row-major. Bt: bf16 [N,K] row-major (i.e. B transposed).
// C = A*B + bias, optional tanh-GELU, optional residual add.
// One wave computes a (16*MS) x (16*NS) tile of C.
// Requires: M % (16*MS) == 0, N % (16*NS) == 0, K % 32 == 0. No bounds checks
// -> straight-line K-loop body: clause of fragment loads, then MS*NS WMMAs.
template<int MS, int NS>
__global__ __launch_bounds__(32)
void k_gemm(const unsigned short* __restrict__ A,
            const unsigned short* __restrict__ Bt,
            const float* __restrict__ bias,
            const float* resid, float* Cf, unsigned short* Cb,
            int N, int K, int act)
{
  int mt  = blockIdx.x * (16 * MS);
  int nt0 = blockIdx.y * (16 * NS);
  int lane = threadIdx.x;
  int l16  = lane & 15;
  int hl   = lane >> 4;

  floatx8 acc[MS][NS];
#pragma unroll
  for (int m = 0; m < MS; ++m)
#pragma unroll
    for (int s = 0; s < NS; ++s)
#pragma unroll
      for (int r = 0; r < 8; ++r) acc[m][s][r] = 0.f;

  for (int k0 = 0; k0 < K; k0 += 32) {
    FragB a[MS];
#pragma unroll
    for (int m = 0; m < MS; ++m) {
      const unsigned short* ap = A + (size_t)(mt + m * 16 + l16) * K + k0;
      a[m].u[0] = *(const uint4*)(ap + 16 * hl);       // K chunk [16h, 16h+8)
      a[m].u[1] = *(const uint4*)(ap + 16 * hl + 8);   // K chunk [16h+8, +8)
    }
    FragB b[NS];
#pragma unroll
    for (int s = 0; s < NS; ++s) {
      const unsigned short* bp = Bt + (size_t)(nt0 + s * 16 + l16) * K + k0 + 16 * hl;
      b[s].u[0] = *(const uint4*)(bp);
      b[s].u[1] = *(const uint4*)(bp + 8);
    }
#pragma unroll
    for (int s = 0; s < NS; ++s)
#pragma unroll
      for (int m = 0; m < MS; ++m)
        acc[m][s] = __builtin_amdgcn_wmma_f32_16x16x32_bf16(
            false, a[m].v, false, b[s].v, (short)0, acc[m][s], false, false);
  }

#pragma unroll
  for (int m = 0; m < MS; ++m)
#pragma unroll
    for (int s = 0; s < NS; ++s) {
      int col = nt0 + s * 16 + l16;
      float bv = bias ? bias[col] : 0.f;
#pragma unroll
      for (int r = 0; r < 8; ++r) {
        int row = mt + m * 16 + r + 8 * hl;            // C layout: m = r + 8*half
        float v = acc[m][s][r] + bv;
        if (act == 1) {                                // tanh-approx GELU
          float t = 0.7978845608028654f * (v + 0.044715f * v * v * v);
          v = 0.5f * v * (1.f + tanhf(t));
        }
        size_t o = (size_t)row * N + col;
        if (resid) v += resid[o];
        if (Cf) Cf[o] = v;
        if (Cb) Cb[o] = f2bf(v);
      }
    }
}

// ---------------- attention (one wave per (b, head, 16-row tile)) ----------
__global__ __launch_bounds__(32)
void k_attn(const unsigned short* __restrict__ Qp,   // [B,H,NPAD,DHP] bf16
            const unsigned short* __restrict__ Kp,   // [B,H,NPAD,DHP] bf16
            const unsigned short* __restrict__ Vpt,  // [B,H,DHP,NPAD] bf16
            unsigned short* __restrict__ O)          // [MTOK,HIDDEN] bf16
{
  __shared__ __align__(16) float          S[16][NPAD];
  __shared__ __align__(16) unsigned short P[16][NPAD];
  __shared__ float red[2][16][2];

  int rt = blockIdx.x, hh = blockIdx.y, b = blockIdx.z;
  int lane = threadIdx.x, l16 = lane & 15, hl = lane >> 4;
  size_t head = (size_t)(b * HEADS + hh);
  const unsigned short* q  = Qp  + head * NPAD * DHP;
  const unsigned short* kk = Kp  + head * NPAD * DHP;
  const unsigned short* vv = Vpt + head * DHP * NPAD;

  FragB qf[3];
  const unsigned short* qrow = q + (size_t)(rt * 16 + l16) * DHP;
#pragma unroll
  for (int ks = 0; ks < 3; ++ks) {
    qf[ks].u[0] = *(const uint4*)(qrow + ks * 32 + 16 * hl);
    qf[ks].u[1] = *(const uint4*)(qrow + ks * 32 + 16 * hl + 8);
  }
  const float scale = 0.11785113019775793f;   // 1/sqrt(72)

  for (int ct = 0; ct < NPAD / 16; ++ct) {    // S = Q * K^T  (B frag = K rows)
    floatx8 acc;
#pragma unroll
    for (int r = 0; r < 8; ++r) acc[r] = 0.f;
    const unsigned short* krow = kk + (size_t)(ct * 16 + l16) * DHP;
#pragma unroll
    for (int ks = 0; ks < 3; ++ks) {
      FragB bf;
      bf.u[0] = *(const uint4*)(krow + ks * 32 + 16 * hl);
      bf.u[1] = *(const uint4*)(krow + ks * 32 + 16 * hl + 8);
      acc = __builtin_amdgcn_wmma_f32_16x16x32_bf16(
          false, qf[ks].v, false, bf.v, (short)0, acc, false, false);
    }
#pragma unroll
    for (int r = 0; r < 8; ++r) S[r + 8 * hl][ct * 16 + l16] = acc[r] * scale;
  }
  __syncthreads();

  {                                            // masked softmax over 282 cols
    int row = l16;
    int c0 = hl * 144;
    int c1 = c0 + 144; if (c1 > NPTS) c1 = NPTS;
    float mx = -3.4e38f;
    for (int c = c0; c < c1; ++c) mx = fmaxf(mx, S[row][c]);
    red[0][row][hl] = mx;
    __syncthreads();
    float m = fmaxf(red[0][row][0], red[0][row][1]);
    float sum = 0.f;
    for (int c = c0; c < c1; ++c) { float e = __expf(S[row][c] - m); S[row][c] = e; sum += e; }
    red[1][row][hl] = sum;
    __syncthreads();
    float inv = 1.f / (red[1][row][0] + red[1][row][1]);
    for (int c = c0; c < c1; ++c) P[row][c] = f2bf(S[row][c] * inv);
    if (hl == 1) for (int c = NPTS; c < NPAD; ++c) P[row][c] = 0;
  }
  __syncthreads();

  for (int ct = 0; ct < DHP / 16; ++ct) {      // O = P * V (V pre-transposed)
    floatx8 acc;
#pragma unroll
    for (int r = 0; r < 8; ++r) acc[r] = 0.f;
    for (int ks = 0; ks < NPAD / 32; ++ks) {
      FragB af, bf;
      af.u[0] = *(const uint4*)&P[l16][ks * 32 + 16 * hl];
      af.u[1] = *(const uint4*)&P[l16][ks * 32 + 16 * hl + 8];
      const unsigned short* vrow = vv + (size_t)(ct * 16 + l16) * NPAD + ks * 32 + 16 * hl;
      bf.u[0] = *(const uint4*)(vrow);
      bf.u[1] = *(const uint4*)(vrow + 8);
      acc = __builtin_amdgcn_wmma_f32_16x16x32_bf16(
          false, af.v, false, bf.v, (short)0, acc, false, false);
    }
    int d = ct * 16 + l16;
#pragma unroll
    for (int r = 0; r < 8; ++r) {
      int t = rt * 16 + r + 8 * hl;
      if (t < NPTS && d < DH)
        O[(size_t)(b * NPTS + t) * HIDDEN + hh * DH + d] = f2bf(acc[r]);
    }
  }
}

// ---------------- support kernels ----------------
__global__ __launch_bounds__(32)
void k_ln(const float* __restrict__ X, unsigned short* __restrict__ Y, int C)
{
  int row = blockIdx.x, lane = threadIdx.x;
  const float* xp = X + (size_t)row * C;
  float s = 0.f, ss = 0.f;
  for (int c = lane; c < C; c += 32) { float v = xp[c]; s += v; ss += v * v; }
  for (int off = 16; off > 0; off >>= 1) { s += __shfl_xor(s, off, 32); ss += __shfl_xor(ss, off, 32); }
  float mean = s / C;
  float inv  = rsqrtf(ss / C - mean * mean + 1e-6f);
  unsigned short* yp = Y + (size_t)row * C;
  for (int c = lane; c < C; c += 32) yp[c] = f2bf((xp[c] - mean) * inv);
}

__global__ void k_cvt(const float* __restrict__ X, unsigned short* __restrict__ Y, size_t n)
{
  size_t i = (size_t)blockIdx.x * blockDim.x + threadIdx.x;
  if (i < n) Y[i] = f2bf(X[i]);
}

// fp32 [K,N] -> bf16 [N,K] (transpose + convert)
__global__ __launch_bounds__(256)
void k_wtrans(const float* __restrict__ W, unsigned short* __restrict__ Wt, int K, int N)
{
  __shared__ float t[32][33];
  int nb = blockIdx.x * 32, kb = blockIdx.y * 32;
  int tx = threadIdx.x, ty = threadIdx.y;
  for (int i = ty; i < 32; i += 8) {
    int kk = kb + i, nn = nb + tx;
    t[i][tx] = (kk < K && nn < N) ? W[(size_t)kk * N + nn] : 0.f;
  }
  __syncthreads();
  for (int i = ty; i < 32; i += 8) {
    int nn = nb + i, kk = kb + tx;
    if (nn < N && kk < K) Wt[(size_t)nn * K + kk] = f2bf(t[tx][i]);
  }
}

__global__ void k_pemb(const float* __restrict__ x, const float* __restrict__ W,
                       const float* __restrict__ b, const float* __restrict__ pos,
                       float* __restrict__ h)
{
  size_t idx = (size_t)blockIdx.x * blockDim.x + threadIdx.x;
  if (idx >= (size_t)MTOK * HIDDEN) return;
  int c = idx % HIDDEN;
  size_t tok = idx / HIDDEN;
  int n = (int)(tok % NPTS);
  const float* xp = x + tok * 3;
  h[idx] = xp[0] * W[c] + xp[1] * W[HIDDEN + c] + xp[2] * W[2 * HIDDEN + c]
         + b[c] + pos[(size_t)n * HIDDEN + c];
}

// reorder qkv bf16 [MTOK,3456] into padded Q/K [B,H,NPAD,DHP] and V^T [B,H,DHP,NPAD]
__global__ void k_repack(const unsigned short* __restrict__ qkv,
                         unsigned short* __restrict__ Qp,
                         unsigned short* __restrict__ Kp,
                         unsigned short* __restrict__ Vpt)
{
  size_t idx = (size_t)blockIdx.x * blockDim.x + threadIdx.x;
  size_t total = (size_t)BQ * HEADS * NPAD * DHP;
  if (idx >= total) return;
  int d = idx % DHP;
  int t = (int)((idx / DHP) % NPAD);
  int h = (int)((idx / ((size_t)DHP * NPAD)) % HEADS);
  int b = (int)(idx / ((size_t)DHP * NPAD * HEADS));
  bool valid = (t < NPTS) && (d < DH);
  unsigned short qv = 0, kv = 0, vv = 0;
  if (valid) {
    size_t src = (size_t)(b * NPTS + t) * QKVN + h * DH + d;
    qv = qkv[src]; kv = qkv[src + HIDDEN]; vv = qkv[src + 2 * HIDDEN];
  }
  Qp[idx] = qv; Kp[idx] = kv;
  Vpt[((size_t)(b * HEADS + h) * DHP + d) * NPAD + t] = vv;
}

// ---------------- VQ kernels ----------------
__global__ void k_vq_init(const float* __restrict__ z, float* f_hat, float* f_rest, float* loss)
{
  size_t idx = (size_t)blockIdx.x * blockDim.x + threadIdx.x;
  if (idx == 0) *loss = 0.f;
  if (idx < (size_t)MTOK * DCB) { f_hat[idx] = 0.f; f_rest[idx] = z[idx]; }
}

__global__ void k_vq_wdown(const float* __restrict__ dist, const int* __restrict__ mp,
                           int nb, float* __restrict__ wdown)
{
  int a = blockIdx.x * blockDim.x + threadIdx.x;
  if (a >= nb) return;
  int ia = mp[a];
  float denom = 0.f;
  for (int n = 0; n < NPTS; ++n) denom += 1.f / dist[n * NPTS + ia];
  float inv = 1.f / denom;
  for (int n = 0; n < NPTS; ++n) wdown[n * nb + a] = (1.f / dist[n * NPTS + ia]) * inv;
}

__global__ void k_vq_down(const float* __restrict__ f_rest, const float* __restrict__ wdown,
                          float* __restrict__ zi, int nb)
{
  size_t idx = (size_t)blockIdx.x * blockDim.x + threadIdx.x;
  if (idx >= (size_t)BQ * nb * DCB) return;
  int c = idx % DCB;
  int a = (int)((idx / DCB) % nb);
  int b = (int)(idx / ((size_t)DCB * nb));
  const float* fp = f_rest + (size_t)b * NPTS * DCB + c;
  float s = 0.f;
  for (int n = 0; n < NPTS; ++n) s += fp[(size_t)n * DCB] * wdown[n * nb + a];
  zi[idx] = s;
}

__global__ void k_vq_assign(const float* __restrict__ zi, const float* __restrict__ emb,
                            int* __restrict__ idxout, int npts)
{
  int p = blockIdx.x * blockDim.x + threadIdx.x;
  if (p >= npts) return;
  const float* zp = zi + (size_t)p * DCB;
  float zloc[DCB];
#pragma unroll
  for (int c = 0; c < DCB; ++c) zloc[c] = zp[c];
  float best = 3.4e38f; int bi = 0;
  for (int kc = 0; kc < KCB; ++kc) {
    const float* ep = emb + (size_t)kc * DCB;
    float d = 0.f;
#pragma unroll 8
    for (int c = 0; c < DCB; ++c) { float df = zloc[c] - ep[c]; d += df * df; }
    if (d < best) { best = d; bi = kc; }
  }
  idxout[p] = bi;
}

__global__ void k_vq_wup(const float* __restrict__ dist, const int* __restrict__ mp,
                         int nb, float* __restrict__ wup)
{
  int n = blockIdx.x * blockDim.x + threadIdx.x;
  if (n >= NPTS) return;
  float denom = 0.f;
  for (int a = 0; a < nb; ++a) denom += 1.f / dist[mp[a] * NPTS + n];
  float inv = 1.f / denom;
  for (int a = 0; a < nb; ++a) wup[(size_t)n * nb + a] = (1.f / dist[mp[a] * NPTS + n]) * inv;
}

__global__ void k_vq_apply(const float* __restrict__ emb, const int* __restrict__ idxbuf,
                           const float* __restrict__ wup, float* f_hat, float* f_rest, int nb)
{
  size_t idx = (size_t)blockIdx.x * blockDim.x + threadIdx.x;
  if (idx >= (size_t)MTOK * DCB) return;
  int c = idx % DCB;
  int n = (int)((idx / DCB) % NPTS);
  int b = (int)(idx / ((size_t)DCB * NPTS));
  const int* ib = idxbuf + b * nb;
  const float* wr = wup + (size_t)n * nb;
  float s = 0.f;
  for (int a = 0; a < nb; ++a) s += emb[(size_t)ib[a] * DCB + c] * wr[a];
  f_hat[idx] += s; f_rest[idx] -= s;
}

__global__ void k_vq_apply_last(const float* __restrict__ emb, const int* __restrict__ idxbuf,
                                float* f_hat, float* f_rest)
{
  size_t idx = (size_t)blockIdx.x * blockDim.x + threadIdx.x;
  if (idx >= (size_t)MTOK * DCB) return;
  int c = idx % DCB;
  size_t p = idx / DCB;
  float v = emb[(size_t)idxbuf[p] * DCB + c];
  f_hat[idx] += v; f_rest[idx] -= v;
}

__global__ __launch_bounds__(256)
void k_vq_loss(const float* __restrict__ f_hat, const float* __restrict__ z,
               float* loss, float coef)
{
  __shared__ float sd[256];
  size_t total = (size_t)MTOK * DCB;
  float s = 0.f;
  for (size_t i = (size_t)blockIdx.x * blockDim.x + threadIdx.x; i < total;
       i += (size_t)gridDim.x * blockDim.x) {
    float d = f_hat[i] - z[i]; s += d * d;
  }
  sd[threadIdx.x] = s; __syncthreads();
  for (int off = 128; off > 0; off >>= 1) {
    if ((int)threadIdx.x < off) sd[threadIdx.x] += sd[threadIdx.x + off];
    __syncthreads();
  }
  if (threadIdx.x == 0) atomicAdd(loss, sd[0] * coef);
}

__global__ void k_final(const unsigned short* __restrict__ lnh, const float* __restrict__ Wf,
                        const float* __restrict__ bf, float* __restrict__ out)
{
  int tok = blockIdx.x * blockDim.x + threadIdx.x;
  if (tok >= MTOK) return;
  const unsigned short* xp = lnh + (size_t)tok * HIDDEN;
  float s0 = bf[0], s1 = bf[1], s2 = bf[2];
  for (int c = 0; c < HIDDEN; ++c) {
    float v = bf2f(xp[c]);
    s0 += v * Wf[c * 3 + 0]; s1 += v * Wf[c * 3 + 1]; s2 += v * Wf[c * 3 + 2];
  }
  out[tok * 3 + 0] = tanhf(s0);
  out[tok * 3 + 1] = tanhf(s1);
  out[tok * 3 + 2] = tanhf(s2);
}

// ---------------- host orchestration ----------------
struct BW {
  const float *fc1_b, *fc1_w, *fc2_b, *fc2_w, *proj_b, *proj_w, *qkv_b, *qkv_w;
};

extern "C" void kernel_launch(void* const* d_in, const int* in_sizes, int n_in,
                              void* d_out, int out_size, void* d_ws, size_t ws_size,
                              hipStream_t stream)
{
  (void)n_in; (void)out_size; (void)ws_size;
  // ---- input map: x, params(tree-flattened alphabetically), dist, multi_points
  const float* x        = (const float*)d_in[0];
  const float* codebook = (const float*)d_in[1];
  BW dec[8], enc[8];
  int p = 2;
  for (int i = 0; i < 8; ++i) {
    dec[i].fc1_b  = (const float*)d_in[p + 0]; dec[i].fc1_w  = (const float*)d_in[p + 1];
    dec[i].fc2_b  = (const float*)d_in[p + 2]; dec[i].fc2_w  = (const float*)d_in[p + 3];
    dec[i].proj_b = (const float*)d_in[p + 4]; dec[i].proj_w = (const float*)d_in[p + 5];
    dec[i].qkv_b  = (const float*)d_in[p + 6]; dec[i].qkv_w  = (const float*)d_in[p + 7];
    p += 8;
  }
  const float* decod_b = (const float*)d_in[p++];
  const float* decod_w = (const float*)d_in[p++];
  for (int i = 0; i < 8; ++i) {
    enc[i].fc1_b  = (const float*)d_in[p + 0]; enc[i].fc1_w  = (const float*)d_in[p + 1];
    enc[i].fc2_b  = (const float*)d_in[p + 2]; enc[i].fc2_w  = (const float*)d_in[p + 3];
    enc[i].proj_b = (const float*)d_in[p + 4]; enc[i].proj_w = (const float*)d_in[p + 5];
    enc[i].qkv_b  = (const float*)d_in[p + 6]; enc[i].qkv_w  = (const float*)d_in[p + 7];
    p += 8;
  }
  const float* encod_b   = (const float*)d_in[p++];
  const float* encod_w   = (const float*)d_in[p++];
  const float* final_b   = (const float*)d_in[p++];
  const float* final_w   = (const float*)d_in[p++];
  const float* p_emb_b   = (const float*)d_in[p++];
  const float* p_emb_w   = (const float*)d_in[p++];
  const float* pos_embed = (const float*)d_in[p++];
  const float* dist      = (const float*)d_in[p++];
  const int* mp[6];
  int mp_base = p;
  for (int i = 0; i < 6; ++i) mp[i] = (const int*)d_in[p++];

  // ---- workspace carve ----
  uintptr_t base = (uintptr_t)d_ws; size_t off = 0;
  auto alloc = [&](size_t bytes) -> void* {
    void* r = (void*)(base + off);
    off = (off + bytes + 255) & ~(size_t)255;
    return r;
  };
  float*          h     = (float*)alloc((size_t)MTOK * HIDDEN * 4);
  unsigned short* lnb   = (unsigned short*)alloc((size_t)MTOK * HIDDEN * 2);
  unsigned short* qkvb  = (unsigned short*)alloc((size_t)MTOK * QKVN * 2);
  unsigned short* Qp    = (unsigned short*)alloc((size_t)BQ * HEADS * NPAD * DHP * 2);
  unsigned short* Kp    = (unsigned short*)alloc((size_t)BQ * HEADS * NPAD * DHP * 2);
  unsigned short* Vpt   = (unsigned short*)alloc((size_t)BQ * HEADS * DHP * NPAD * 2);
  unsigned short* Ob    = (unsigned short*)alloc((size_t)MTOK * HIDDEN * 2);
  unsigned short* m1b   = (unsigned short*)alloc((size_t)MTOK * MLPH * 2);
  unsigned short* qkvT  = (unsigned short*)alloc((size_t)HIDDEN * QKVN * 2);
  unsigned short* projT = (unsigned short*)alloc((size_t)HIDDEN * HIDDEN * 2);
  unsigned short* fc1T  = (unsigned short*)alloc((size_t)HIDDEN * MLPH * 2);
  unsigned short* fc2T  = (unsigned short*)alloc((size_t)MLPH * HIDDEN * 2);
  unsigned short* encT  = (unsigned short*)alloc((size_t)DCB * HIDDEN * 2);
  unsigned short* decT  = (unsigned short*)alloc((size_t)HIDDEN * DCB * 2);
  float* z      = (float*)alloc((size_t)MTOK * DCB * 4);
  float* f_hat  = (float*)alloc((size_t)MTOK * DCB * 4);
  float* f_rest = (float*)alloc((size_t)MTOK * DCB * 4);
  float* zi     = (float*)alloc((size_t)BQ * 128 * DCB * 4);
  float* wdown  = (float*)alloc((size_t)NPTS * 128 * 4);
  float* wup    = (float*)alloc((size_t)NPTS * 128 * 4);
  int*   idxbuf = (int*)alloc((size_t)MTOK * 4);
  unsigned short* fhb = (unsigned short*)alloc((size_t)MTOK * DCB * 2);

  float* out      = (float*)d_out;
  float* loss_ptr = out + (size_t)MTOK * 3;

  dim3 tb(32, 8);
  auto run_block = [&](const BW& w) {
    k_wtrans<<<dim3((QKVN + 31) / 32, (HIDDEN + 31) / 32), tb, 0, stream>>>(w.qkv_w, qkvT, HIDDEN, QKVN);
    k_wtrans<<<dim3((HIDDEN + 31) / 32, (HIDDEN + 31) / 32), tb, 0, stream>>>(w.proj_w, projT, HIDDEN, HIDDEN);
    k_wtrans<<<dim3((MLPH + 31) / 32, (HIDDEN + 31) / 32), tb, 0, stream>>>(w.fc1_w, fc1T, HIDDEN, MLPH);
    k_wtrans<<<dim3((HIDDEN + 31) / 32, (MLPH + 31) / 32), tb, 0, stream>>>(w.fc2_w, fc2T, MLPH, HIDDEN);

    k_ln<<<MTOK, 32, 0, stream>>>(h, lnb, HIDDEN);
    k_gemm<2, 8><<<dim3(MTOK / 32, QKVN / 128), 32, 0, stream>>>(
        lnb, qkvT, w.qkv_b, nullptr, nullptr, qkvb, QKVN, HIDDEN, 0);
    {
      size_t tot = (size_t)BQ * HEADS * NPAD * DHP;
      k_repack<<<(unsigned)((tot + 255) / 256), 256, 0, stream>>>(qkvb, Qp, Kp, Vpt);
    }
    k_attn<<<dim3(NPAD / 16, HEADS, BQ), 32, 0, stream>>>(Qp, Kp, Vpt, Ob);
    k_gemm<2, 8><<<dim3(MTOK / 32, HIDDEN / 128), 32, 0, stream>>>(
        Ob, projT, w.proj_b, h, h, nullptr, HIDDEN, HIDDEN, 0);
    k_ln<<<MTOK, 32, 0, stream>>>(h, lnb, HIDDEN);
    k_gemm<2, 8><<<dim3(MTOK / 32, MLPH / 128), 32, 0, stream>>>(
        lnb, fc1T, w.fc1_b, nullptr, nullptr, m1b, MLPH, HIDDEN, 1);
    k_gemm<2, 8><<<dim3(MTOK / 32, HIDDEN / 128), 32, 0, stream>>>(
        m1b, fc2T, w.fc2_b, h, h, nullptr, HIDDEN, MLPH, 0);
  };

  // ---- patch embed + pos embed ----
  {
    size_t tot = (size_t)MTOK * HIDDEN;
    k_pemb<<<(unsigned)((tot + 255) / 256), 256, 0, stream>>>(x, p_emb_w, p_emb_b, pos_embed, h);
  }
  // ---- encoder ----
  for (int i = 0; i < 8; ++i) run_block(enc[i]);
  // ---- encode to latent ----
  k_wtrans<<<dim3((DCB + 31) / 32, (HIDDEN + 31) / 32), tb, 0, stream>>>(encod_w, encT, HIDDEN, DCB);
  {
    size_t tot = (size_t)MTOK * HIDDEN;
    k_cvt<<<(unsigned)((tot + 255) / 256), 256, 0, stream>>>(h, lnb, tot);
  }
  k_gemm<2, 4><<<dim3(MTOK / 32, DCB / 64), 32, 0, stream>>>(
      lnb, encT, encod_b, nullptr, z, nullptr, DCB, HIDDEN, 0);

  // ---- multi-scale VQ ----
  {
    size_t tot = (size_t)MTOK * DCB;
    k_vq_init<<<(unsigned)((tot + 255) / 256), 256, 0, stream>>>(z, f_hat, f_rest, loss_ptr);
  }
  float coef = 1.25f / (6.f * (float)MTOK * (float)DCB);   // (1+beta)/(S * numel)
  for (int i = 0; i < 6; ++i) {
    int nb = in_sizes[mp_base + i];
    if (i < 5) {
      k_vq_wdown<<<1, 128, 0, stream>>>(dist, mp[i], nb, wdown);
      {
        size_t tot = (size_t)BQ * nb * DCB;
        k_vq_down<<<(unsigned)((tot + 255) / 256), 256, 0, stream>>>(f_rest, wdown, zi, nb);
      }
      k_vq_assign<<<(BQ * nb + 63) / 64, 64, 0, stream>>>(zi, codebook, idxbuf, BQ * nb);
      k_vq_wup<<<(NPTS + 63) / 64, 64, 0, stream>>>(dist, mp[i], nb, wup);
      {
        size_t tot = (size_t)MTOK * DCB;
        k_vq_apply<<<(unsigned)((tot + 255) / 256), 256, 0, stream>>>(
            codebook, idxbuf, wup, f_hat, f_rest, nb);
      }
    } else {
      k_vq_assign<<<(MTOK + 63) / 64, 64, 0, stream>>>(f_rest, codebook, idxbuf, MTOK);
      {
        size_t tot = (size_t)MTOK * DCB;
        k_vq_apply_last<<<(unsigned)((tot + 255) / 256), 256, 0, stream>>>(
            codebook, idxbuf, f_hat, f_rest);
      }
    }
    k_vq_loss<<<256, 256, 0, stream>>>(f_hat, z, loss_ptr, coef);
  }

  // ---- decode ----
  k_wtrans<<<dim3((HIDDEN + 31) / 32, (DCB + 31) / 32), tb, 0, stream>>>(decod_w, decT, DCB, HIDDEN);
  {
    size_t tot = (size_t)MTOK * DCB;
    k_cvt<<<(unsigned)((tot + 255) / 256), 256, 0, stream>>>(f_hat, fhb, tot);
  }
  k_gemm<2, 8><<<dim3(MTOK / 32, HIDDEN / 128), 32, 0, stream>>>(
      fhb, decT, decod_b, nullptr, h, nullptr, HIDDEN, DCB, 0);
  for (int i = 0; i < 8; ++i) run_block(dec[i]);

  // ---- final head ----
  k_ln<<<MTOK, 32, 0, stream>>>(h, lnb, HIDDEN);
  k_final<<<(MTOK + 255) / 256, 256, 0, stream>>>(lnb, final_w, final_b, out);
}
